// ESN_58317065945127
// MI455X (gfx1250) — compile-verified, matrix operand order checked by reading, not run on hardware
//
#include <hip/hip_runtime.h>

typedef unsigned short u16;
typedef __attribute__((ext_vector_type(16))) __bf16 v16bf;
typedef __attribute__((ext_vector_type(8)))  float  v8f;

union FragBF { v16bf v; unsigned u[8]; };

#define T_STEPS 2048
#define BATCH   128
#define HID     1024
#define DIN     128
#define LASTN   20
#define TAIL0   (T_STEPS - LASTN)
#define KTOT    (HID + DIN)      // 1152 : concatenated K for [h | x_t] @ [w_r ; w_in]
#define LDS_PITCH 1160           // multiple of 8 (16B-aligned rows), 4-bank skew per col

__device__ __forceinline__ u16 f2bf(float f) {
  unsigned u = __float_as_uint(f);
  unsigned r = u + 0x7FFFu + ((u >> 16) & 1u);   // round-to-nearest-even
  return (u16)(r >> 16);
}

__device__ __forceinline__ unsigned f2bf2(float a, float b) {
  return (unsigned)f2bf(a) | ((unsigned)f2bf(b) << 16);
}

__device__ __forceinline__ float fast_tanh(float x) {
  // tanh(x) = 1 - 2/(e^{2x}+1);  e^{2x} = exp2(2*log2(e)*x). TRANS ops co-execute with WMMA.
  float e = __builtin_amdgcn_exp2f(x * 2.8853900817779268f);
  return 1.0f - 2.0f * __builtin_amdgcn_rcpf(e + 1.0f);
}

__device__ __forceinline__ void frag_set(FragBF& f, uint4 lo, uint4 hi) {
  f.u[0] = lo.x; f.u[1] = lo.y; f.u[2] = lo.z; f.u[3] = lo.w;
  f.u[4] = hi.x; f.u[5] = hi.y; f.u[6] = hi.z; f.u[7] = hi.w;
}

// ---------------------------------------------------------------------------
// Kernel 1: fp32 -> bf16 (paired stores), and wcat = [w_r ; w_in] (1152 x 1024)
// ---------------------------------------------------------------------------
__global__ void esn_cvt(const float* __restrict__ x, const float* __restrict__ w_in,
                        const float* __restrict__ w_r,
                        unsigned* __restrict__ xbf2, unsigned* __restrict__ wcat2) {
  long i0 = (long)blockIdx.x * blockDim.x + threadIdx.x;
  long stride = (long)gridDim.x * blockDim.x;
  const float2* xf2 = (const float2*)x;
  const long NX2 = (long)BATCH * T_STEPS * DIN / 2;
  for (long i = i0; i < NX2; i += stride) {
    float2 v = xf2[i];
    xbf2[i] = f2bf2(v.x, v.y);
  }
  const long NW2 = (long)KTOT * HID / 2;   // pairs never cross a row (HID even)
  for (long i = i0; i < NW2; i += stride) {
    long e0 = i * 2;
    long r = e0 >> 10, c = e0 & (HID - 1);
    float v0, v1;
    if (r < HID) { v0 = w_r[r * HID + c];         v1 = w_r[r * HID + c + 1]; }
    else         { v0 = w_in[(r - HID) * HID + c]; v1 = w_in[(r - HID) * HID + c + 1]; }
    wcat2[i] = f2bf2(v0, v1);
  }
}

// ---------------------------------------------------------------------------
// Kernel 2: persistent recurrent scan. 16 blocks x 256 threads (8 waves).
// Block g owns cols [g*64, g*64+64); its slice of [w_r;w_in] (1152x64 bf16,
// transposed) is LDS-resident for all 2048 steps.
// Wave w = (row-strip rs = w>>1, col-half ch = w&1): computes 32 rows x 32 cols
// as a 2x2 grid of 16x16 WMMA tiles -> each loaded fragment feeds 2 WMMAs,
// and all 4 frags are loaded before the 4 independent WMMAs (one wait/burst).
// One agent-scope barrier per step; h double-buffered (fp32 master + bf16).
// ---------------------------------------------------------------------------
__global__ __launch_bounds__(256, 1) void esn_scan(
    const u16* __restrict__ xbf, const u16* __restrict__ wcat,
    float* __restrict__ hf, u16* __restrict__ hbf,
    float* __restrict__ hs, unsigned* __restrict__ cnt) {
  extern __shared__ u16 ldsW[];   // [64][LDS_PITCH] : Wt[n_local][k]
  const int tid   = threadIdx.x;
  const int lane  = tid & 31;
  const int wave  = tid >> 5;
  const int rs    = wave >> 1;                  // row strip: 32 rows
  const int ch    = wave & 1;                   // col half : 32 cols
  const int half8 = (lane >> 4) * 8;            // K-offset of this wave-half
  const int rowA0 = rs * 32 + (lane & 15);      // A rows for the two row tiles
  const int rowA1 = rowA0 + 16;
  const int nl0   = ch * 32 + (lane & 15);      // B cols for the two col tiles
  const int nl1   = nl0 + 16;
  const int g     = blockIdx.x;

  // Stage this block's 64-column slice of wcat into LDS (transposed, padded)
  for (int idx = tid; idx < 64 * KTOT; idx += 256) {
    int k = idx >> 6, nl = idx & 63;
    ldsW[nl * LDS_PITCH + k] = wcat[(long)k * HID + g * 64 + nl];
  }
  __syncthreads();

  const uint4* lds4 = reinterpret_cast<const uint4*>(ldsW);
  const uint4* xp4  = reinterpret_cast<const uint4*>(xbf);
  const int bbase0 = nl0 * (LDS_PITCH >> 1);    // dword offsets of B rows in LDS
  const int bbase1 = nl1 * (LDS_PITCH >> 1);

  for (int t = 0; t < T_STEPS; ++t) {
    const int cur = t & 1, prev = cur ^ 1;
    v8f a00 = {}, a01 = {}, a10 = {}, a11 = {};   // acc[rowTile][colTile]

    if (t + 1 < T_STEPS) {  // pull next timestep's x rows toward L2 (global_prefetch)
      __builtin_prefetch(xbf + ((long)rowA0 * T_STEPS + t + 1) * DIN, 0, 0);
      __builtin_prefetch(xbf + ((long)rowA1 * T_STEPS + t + 1) * DIN, 0, 0);
    }

    // ---- recurrent part: h(t-1) @ w_r  (skipped at t==0) ----
    if (t > 0) {
      const uint4* hp4 = reinterpret_cast<const uint4*>(hbf + (long)prev * BATCH * HID);
      for (int kb = 0; kb < HID; kb += 32) {
        const int koff = (kb + half8) >> 1;                       // dwords
        const long di0 = ((long)rowA0 * (HID / 2) + koff) >> 2;   // uint4 idx, 16B aligned
        const long di1 = ((long)rowA1 * (HID / 2) + koff) >> 2;
        const int  bi0 = (bbase0 + koff) >> 2;
        const int  bi1 = (bbase1 + koff) >> 2;
        FragBF A0, A1, B0, B1;
        frag_set(A0, hp4[di0], hp4[di0 + 2]);
        frag_set(A1, hp4[di1], hp4[di1 + 2]);
        frag_set(B0, lds4[bi0], lds4[bi0 + 2]);
        frag_set(B1, lds4[bi1], lds4[bi1 + 2]);
        a00 = __builtin_amdgcn_wmma_f32_16x16x32_bf16(false, A0.v, false, B0.v, (short)0, a00, false, false);
        a01 = __builtin_amdgcn_wmma_f32_16x16x32_bf16(false, A0.v, false, B1.v, (short)0, a01, false, false);
        a10 = __builtin_amdgcn_wmma_f32_16x16x32_bf16(false, A1.v, false, B0.v, (short)0, a10, false, false);
        a11 = __builtin_amdgcn_wmma_f32_16x16x32_bf16(false, A1.v, false, B1.v, (short)0, a11, false, false);
      }
    }

    // ---- input part: x_t @ w_in (K = 1024..1151 of the concatenated slice) ----
    {
      const long xdw0 = ((long)rowA0 * T_STEPS + t) * (DIN / 2);  // dword bases
      const long xdw1 = ((long)rowA1 * T_STEPS + t) * (DIN / 2);
      for (int kb = 0; kb < DIN; kb += 32) {
        const int koff = (kb + half8) >> 1;
        const long di0 = (xdw0 + koff) >> 2;
        const long di1 = (xdw1 + koff) >> 2;
        const int  bi0 = (bbase0 + ((HID + kb + half8) >> 1)) >> 2;
        const int  bi1 = (bbase1 + ((HID + kb + half8) >> 1)) >> 2;
        FragBF A0, A1, B0, B1;
        frag_set(A0, xp4[di0], xp4[di0 + 2]);
        frag_set(A1, xp4[di1], xp4[di1 + 2]);
        frag_set(B0, lds4[bi0], lds4[bi0 + 2]);
        frag_set(B1, lds4[bi1], lds4[bi1 + 2]);
        a00 = __builtin_amdgcn_wmma_f32_16x16x32_bf16(false, A0.v, false, B0.v, (short)0, a00, false, false);
        a01 = __builtin_amdgcn_wmma_f32_16x16x32_bf16(false, A0.v, false, B1.v, (short)0, a01, false, false);
        a10 = __builtin_amdgcn_wmma_f32_16x16x32_bf16(false, A1.v, false, B0.v, (short)0, a10, false, false);
        a11 = __builtin_amdgcn_wmma_f32_16x16x32_bf16(false, A1.v, false, B1.v, (short)0, a11, false, false);
      }
    }

    // ---- epilogue: tanh + leaky mix, write fp32 master + bf16 copy (+ tail) ----
    float*       hfc = hf  + (long)cur  * BATCH * HID;
    const float* hfp = hf  + (long)prev * BATCH * HID;
    u16*         hbc = hbf + (long)cur  * BATCH * HID;
    v8f* accp[2][2] = { { &a00, &a01 }, { &a10, &a11 } };
#pragma unroll
    for (int ai = 0; ai < 2; ++ai) {
#pragma unroll
      for (int nt = 0; nt < 2; ++nt) {
        const int gcol = g * 64 + ch * 32 + nt * 16 + (lane & 15);
        const int mb   = rs * 32 + ai * 16 + half8;   // D: VGPR j -> row mb+j
#pragma unroll
        for (int j = 0; j < 8; ++j) {
          int m = mb + j;
          float hh   = fast_tanh((*accp[ai][nt])[j]);
          float hnew = (t == 0) ? hh : 0.1f * hfp[m * HID + gcol] + 0.9f * hh;
          hfc[m * HID + gcol] = hnew;
          hbc[m * HID + gcol] = f2bf(hnew);
          if (t >= TAIL0)
            hs[((long)(t - TAIL0) * BATCH + m) * HID + gcol] = hnew;
        }
      }
    }

    // ---- one grid-wide barrier per step (monotonic agent-scope counter) ----
    __syncthreads();
    if (tid == 0) {
      __hip_atomic_fetch_add(cnt, 1u, __ATOMIC_ACQ_REL, __HIP_MEMORY_SCOPE_AGENT);
      unsigned target = 16u * (unsigned)(t + 1);
      while (__hip_atomic_load(cnt, __ATOMIC_ACQUIRE, __HIP_MEMORY_SCOPE_AGENT) < target)
        __builtin_amdgcn_s_sleep(1);
    }
    __syncthreads();
  }
}

// ---------------------------------------------------------------------------
// Kernel 3: out[b,o] = bias[o] + sum_{i,h} hs[i,b,h] * W[i*H+h, o]
// ---------------------------------------------------------------------------
__global__ void esn_out(const float* __restrict__ hs, const float* __restrict__ W,
                        const float* __restrict__ bias, float* __restrict__ out) {
  int b = blockIdx.x;
  int o = threadIdx.x;
  float acc = bias[o];
  for (int i = 0; i < LASTN; ++i) {
    const float* hrow  = hs + ((long)i * BATCH + b) * HID;
    const float* wbase = W  + (long)i * HID * 32;
    for (int hcol = 0; hcol < HID; ++hcol)
      acc = fmaf(hrow[hcol], wbase[hcol * 32 + o], acc);
  }
  out[b * 32 + o] = acc;
}

extern "C" void kernel_launch(void* const* d_in, const int* in_sizes, int n_in,
                              void* d_out, int out_size, void* d_ws, size_t ws_size,
                              hipStream_t stream) {
  (void)in_sizes; (void)n_in; (void)out_size; (void)ws_size;
  const float* x    = (const float*)d_in[0];   // [128,2048,128]
  const float* w_in = (const float*)d_in[1];   // [128,1024]
  const float* w_r  = (const float*)d_in[2];   // [1024,1024]
  const float* W    = (const float*)d_in[3];   // [20480,32]
  const float* bias = (const float*)d_in[4];   // [32]

  char* ws = (char*)d_ws;
  size_t off = 0;
  unsigned* cnt = (unsigned*)(ws + off); off += 256;
  u16*  xbf  = (u16*) (ws + off); off += (size_t)BATCH * T_STEPS * DIN * 2; // 64 MiB
  u16*  wcat = (u16*) (ws + off); off += (size_t)KTOT * HID * 2;            // 2.25 MiB
  float* hf  = (float*)(ws + off); off += (size_t)2 * BATCH * HID * 4;      // 1 MiB
  u16*  hbf  = (u16*) (ws + off); off += (size_t)2 * BATCH * HID * 2;       // 0.5 MiB
  float* hs  = (float*)(ws + off); off += (size_t)LASTN * BATCH * HID * 4;  // 10 MiB

  const int ldsBytes = 64 * LDS_PITCH * 2;  // 145 KiB (of 320 KiB/WGP)
  (void)hipFuncSetAttribute((const void*)esn_scan,
                            hipFuncAttributeMaxDynamicSharedMemorySize, ldsBytes);

  (void)hipMemsetAsync(cnt, 0, 256, stream);                 // reset barrier counter
  esn_cvt <<<1024, 256, 0, stream>>>(x, w_in, w_r, (unsigned*)xbf, (unsigned*)wcat);
  esn_scan<<<16, 256, ldsBytes, stream>>>(xbf, wcat, hf, hbf, hs, cnt);
  esn_out <<<BATCH, 32, 0, stream>>>(hs, W, bias, (float*)d_out);
}
